// GGNNEncoder_12232066859183
// MI455X (gfx1250) — matrix operand order repaired and checked by reading, not compiled, not run on hardware
//
#include <hip/hip_runtime.h>
#include <math.h>

// ---------------------------------------------------------------------------
// GGNN encoder for MI455X (gfx1250): all GEMMs on v_wmma_f32_16x16x32_bf16.
// ---------------------------------------------------------------------------

typedef __bf16 bf16_t;
typedef __attribute__((ext_vector_type(16))) __bf16 v16bf;
typedef __attribute__((ext_vector_type(8)))  __bf16 v8bf;
typedef __attribute__((ext_vector_type(8)))  float  v8f;

#define DD     256          // hidden dim
#define EE     4            // edge types
#define ATOMS  128
#define MBATCH 1024
#define MM     512          // readout dim
#define ROWS   (MBATCH * ATOMS)   // 131072 flattened node rows

// ---- scalar helpers -------------------------------------------------------

// Let the backend pick the native bf16 convert (v_cvt_*bf16_f32 on gfx1250).
__device__ __forceinline__ bf16_t f2bf(float f) { return (bf16_t)f; }
__device__ __forceinline__ float  bf2f(bf16_t b) { return (float)b; }

__device__ __forceinline__ float sigmoidf_(float x) { return 1.0f / (1.0f + __expf(-x)); }

__device__ __forceinline__ v8f v8f_splat(float x) {
  v8f r;
#pragma unroll
  for (int i = 0; i < 8; ++i) r[i] = x;
  return r;
}

__device__ __forceinline__ v8f wmma_bf(v16bf a, v16bf b, v8f c) {
  // D = A(16x32) * B(32x16) + C, f32 accumulate
  return __builtin_amdgcn_wmma_f32_16x16x32_bf16(false, a, false, b, (short)0, c, false, false);
}

// ---- WMMA fragment loaders (per ISA 7.12.2 16-bit layouts) ----------------
// A 16x32: lane holds row M=lane%16; K chunks [kb, kb+8) and [kb+16, kb+24)
// where kb = k0 + (lane>=16 ? 8 : 0).  Source row-major, bf16.
__device__ __forceinline__ v16bf ldA_bf(const bf16_t* __restrict__ rowp, int k0, int lane) {
  const bf16_t* p = rowp + k0 + ((lane & 16) ? 8 : 0);
  v8bf lo = *(const v8bf*)(p);
  v8bf hi = *(const v8bf*)(p + 16);
  v16bf a;
#pragma unroll
  for (int i = 0; i < 8; ++i) { a[i] = lo[i]; a[i + 8] = hi[i]; }
  return a;
}

// Same A layout but fp32 source, convert on load (used for adj).
__device__ __forceinline__ v16bf ldA_f32(const float* __restrict__ rowp, int k0, int lane) {
  const float* p = rowp + k0 + ((lane & 16) ? 8 : 0);
  float4 c0 = *(const float4*)(p);
  float4 c1 = *(const float4*)(p + 4);
  float4 c2 = *(const float4*)(p + 16);
  float4 c3 = *(const float4*)(p + 20);
  v16bf a;
  a[0]  = f2bf(c0.x); a[1]  = f2bf(c0.y); a[2]  = f2bf(c0.z); a[3]  = f2bf(c0.w);
  a[4]  = f2bf(c1.x); a[5]  = f2bf(c1.y); a[6]  = f2bf(c1.z); a[7]  = f2bf(c1.w);
  a[8]  = f2bf(c2.x); a[9]  = f2bf(c2.y); a[10] = f2bf(c2.z); a[11] = f2bf(c2.w);
  a[12] = f2bf(c3.x); a[13] = f2bf(c3.y); a[14] = f2bf(c3.z); a[15] = f2bf(c3.w);
  return a;
}

// B 32x16: lane holds column N=lane%16; K chunks [kb, kb+8) and [kb+16, kb+24)
// where kb = k0 + (lane>=16 ? 8 : 0).  rowp = base of the weight row (= output column),
// row-major [N][K].
__device__ __forceinline__ v16bf ldB_bf(const bf16_t* __restrict__ rowp, int k0, int lane) {
  const bf16_t* p = rowp + k0 + ((lane & 16) ? 8 : 0);
  v8bf lo = *(const v8bf*)(p);
  v8bf hi = *(const v8bf*)(p + 16);
  v16bf b;
#pragma unroll
  for (int i = 0; i < 8; ++i) { b[i] = lo[i]; b[i + 8] = hi[i]; }
  return b;
}

// ---- kernel 0: fp32 -> bf16 weight conversion -----------------------------
__global__ void k_convert(const float* __restrict__ src, bf16_t* __restrict__ dst, int n) {
  int i = blockIdx.x * blockDim.x + threadIdx.x;
  if (i < n) dst[i] = f2bf(src[i]);
}

// ---- kernel 1: embedding gather -> bf16 h and h0 --------------------------
__global__ void k_embed(const int* __restrict__ atom, const float* __restrict__ embed,
                        bf16_t* __restrict__ h, bf16_t* __restrict__ h0) {
  int i = blockIdx.x * blockDim.x + threadIdx.x;   // over ROWS*DD
  int row = i >> 8;
  int d   = i & 255;
  int a   = atom[row];
  bf16_t v = f2bf(embed[a * DD + d]);
  h[i]  = v;
  h0[i] = v;
}

// ---- kernel 2: fused message GEMM + adjacency bmm (per batch element) -----
// Per block (8 waves, batch b):
//   for each edge type e:
//     phase 1: m_e = h[b] @ msgW_e^T + b_e  (128x256) written TRANSPOSED to
//              LDS mT[d][j] (bf16, 64KB) so phase-2 B fragments are contiguous.
//     phase 2: msum[b] += bf16(adj[b,e]) @ m_e   (f32 accumulators in VGPRs)
__global__ __launch_bounds__(256) void k_msg(const bf16_t* __restrict__ hcur,
                                             const float*  __restrict__ adj,
                                             const bf16_t* __restrict__ wmsg,  // [1024][256] bf16
                                             const float*  __restrict__ msgb,  // [1024]
                                             bf16_t* __restrict__ msum) {
  __shared__ bf16_t mT[256 * 128];   // [d_local][j], 65536 bytes

  const int b    = blockIdx.x;
  const int lane = threadIdx.x & 31;
  const int w    = threadIdx.x >> 5;   // wave id 0..7 -> 16-row strip
  const int lrow = lane & 15;

  // A fragments of h[b] for this wave's row strip (invariant across e).
  const bf16_t* hrow = hcur + ((size_t)b * ATOMS + w * 16 + lrow) * DD;
  v16bf aH[8];
#pragma unroll
  for (int k = 0; k < 8; ++k) aH[k] = ldA_bf(hrow, k * 32, lane);

  v8f accS[16];
#pragma unroll
  for (int t = 0; t < 16; ++t) accS[t] = v8f_splat(0.0f);

  for (int e = 0; e < EE; ++e) {
    // ---- phase 1: message slice for edge e -> transposed LDS ----
#pragma unroll 1
    for (int nt = 0; nt < 16; ++nt) {
      const int wr = e * 256 + nt * 16 + lrow;     // weight row = output column
      v8f c = v8f_splat(msgb[wr]);                 // bias broadcast down column
      const bf16_t* wrp = wmsg + (size_t)wr * DD;
#pragma unroll
      for (int k = 0; k < 8; ++k) c = wmma_bf(aH[k], ldB_bf(wrp, k * 32, lane), c);
      // Transposed store: D-frag lane owns col d = nt*16+lrow, rows j = strip
      // base + v (+8 for upper half-wave) -> 8 consecutive j -> one 16B store.
      const int dcol  = nt * 16 + lrow;
      const int jbase = w * 16 + ((lane & 16) ? 8 : 0);
      v8bf pk;
#pragma unroll
      for (int v = 0; v < 8; ++v) pk[v] = f2bf(c[v]);
      *(v8bf*)(&mT[dcol * 128 + jbase]) = pk;
    }
    __syncthreads();

    // ---- phase 2: accS += adj[b,e] (rows = this strip) @ m_e ----
    const float* adjrow =
        adj + (((size_t)b * EE + e) * ATOMS + w * 16 + lrow) * ATOMS;
#pragma unroll 1
    for (int kt = 0; kt < 4; ++kt) {             // K = atoms j, 4 x 32
      if (kt < 3) __builtin_prefetch(adjrow + (kt + 1) * 32, 0, 0);
      v16bf aA = ldA_f32(adjrow, kt * 32, lane);
#pragma unroll
      for (int nt = 0; nt < 16; ++nt) {
        const bf16_t* mp = &mT[(nt * 16 + lrow) * 128];
        accS[nt] = wmma_bf(aA, ldB_bf(mp, kt * 32, lane), accS[nt]);
      }
    }
    __syncthreads();
  }

  // store msum strip (bf16, row-major) for the GRU stage
  bf16_t* mout = msum + (size_t)b * ATOMS * DD;
#pragma unroll
  for (int nt = 0; nt < 16; ++nt) {
    const int col = nt * 16 + lrow;
#pragma unroll
    for (int v = 0; v < 8; ++v) {
      const int row = w * 16 + v + ((lane & 16) ? 8 : 0);
      mout[(size_t)row * DD + col] = f2bf(accS[nt][v]);
    }
  }
}

// ---- kernel 3: fused GRU cell ---------------------------------------------
// Block = one 16-row strip; wave w owns column tiles 2w, 2w+1 of D=256.
// Six f32 WMMA accumulators per tile (r/z/n gates x input/hidden GEMM).
__global__ __launch_bounds__(256) void k_gru(const bf16_t* __restrict__ msum,
                                             const bf16_t* __restrict__ hin,
                                             const bf16_t* __restrict__ wih,  // [768][256] bf16
                                             const float*  __restrict__ bih,
                                             const bf16_t* __restrict__ whh,  // [768][256] bf16
                                             const float*  __restrict__ bhh,
                                             bf16_t* __restrict__ hout, int hasH) {
  const int lane = threadIdx.x & 31;
  const int w    = threadIdx.x >> 5;
  const int lrow = lane & 15;
  const size_t r0 = (size_t)blockIdx.x * 16;

  const bf16_t* mrow = msum + (r0 + lrow) * DD;
  const bf16_t* hrow = hin  + (r0 + lrow) * DD;

  v8f aIR[2], aIZ[2], aIN[2], aHR[2], aHZ[2], aHN[2];
#pragma unroll
  for (int t = 0; t < 2; ++t) {
    aIR[t] = v8f_splat(0.0f); aIZ[t] = v8f_splat(0.0f); aIN[t] = v8f_splat(0.0f);
    aHR[t] = v8f_splat(0.0f); aHZ[t] = v8f_splat(0.0f); aHN[t] = v8f_splat(0.0f);
  }

#pragma unroll 1
  for (int k = 0; k < 8; ++k) {                    // K = 256
    v16bf am = ldA_bf(mrow, k * 32, lane);
    v16bf ah = am;                                  // overwritten if hasH
    if (hasH) ah = ldA_bf(hrow, k * 32, lane);
#pragma unroll
    for (int t = 0; t < 2; ++t) {
      const int n0 = w * 32 + t * 16;
      aIR[t] = wmma_bf(am, ldB_bf(wih + (size_t)(      n0 + lrow) * DD, k * 32, lane), aIR[t]);
      aIZ[t] = wmma_bf(am, ldB_bf(wih + (size_t)(256 + n0 + lrow) * DD, k * 32, lane), aIZ[t]);
      aIN[t] = wmma_bf(am, ldB_bf(wih + (size_t)(512 + n0 + lrow) * DD, k * 32, lane), aIN[t]);
      if (hasH) {
        aHR[t] = wmma_bf(ah, ldB_bf(whh + (size_t)(      n0 + lrow) * DD, k * 32, lane), aHR[t]);
        aHZ[t] = wmma_bf(ah, ldB_bf(whh + (size_t)(256 + n0 + lrow) * DD, k * 32, lane), aHZ[t]);
        aHN[t] = wmma_bf(ah, ldB_bf(whh + (size_t)(512 + n0 + lrow) * DD, k * 32, lane), aHN[t]);
      }
    }
  }

#pragma unroll
  for (int t = 0; t < 2; ++t) {
    const int col = w * 32 + t * 16 + lrow;
    const float bir = bih[col], biz = bih[256 + col], bin_ = bih[512 + col];
    const float bhr = bhh[col], bhz = bhh[256 + col], bhn  = bhh[512 + col];
#pragma unroll
    for (int v = 0; v < 8; ++v) {
      const size_t row = r0 + v + ((lane & 16) ? 8 : 0);
      const float hr = hasH ? aHR[t][v] : 0.0f;
      const float hz = hasH ? aHZ[t][v] : 0.0f;
      const float hn = hasH ? aHN[t][v] : 0.0f;
      const float hp = hasH ? bf2f(hin[row * DD + col]) : 0.0f;
      const float rg = sigmoidf_(aIR[t][v] + bir + hr + bhr);
      const float zg = sigmoidf_(aIZ[t][v] + biz + hz + bhz);
      const float ng = tanhf(aIN[t][v] + bin_ + rg * (hn + bhn));
      hout[row * DD + col] = f2bf((1.0f - zg) * ng + zg * hp);
    }
  }
}

// ---- kernel 4: fused gated readout ----------------------------------------
// One block per batch element. gate = sigmoid([h,h0] @ iW^T + ib) (K=512),
// val = h @ jW^T + jb (K=256); out[b,:] = sum_atoms gate*val.
__global__ __launch_bounds__(256) void k_readout(const bf16_t* __restrict__ h,
                                                 const bf16_t* __restrict__ h0,
                                                 const bf16_t* __restrict__ wi,  // [512][512] bf16
                                                 const float*  __restrict__ ib,
                                                 const bf16_t* __restrict__ wj,  // [512][256] bf16
                                                 const float*  __restrict__ jb,
                                                 float* __restrict__ out) {
  const int b    = blockIdx.x;
  const int lane = threadIdx.x & 31;
  const int w    = threadIdx.x >> 5;   // wave owns 64 output cols
  const int lrow = lane & 15;

  float psum[4] = {0.0f, 0.0f, 0.0f, 0.0f};

#pragma unroll 1
  for (int s = 0; s < 8; ++s) {                    // atom strips of 16
    const bf16_t* hrow  = h  + ((size_t)b * ATOMS + s * 16 + lrow) * DD;
    const bf16_t* h0row = h0 + ((size_t)b * ATOMS + s * 16 + lrow) * DD;
    v16bf aH[8], aH0[8];
#pragma unroll
    for (int k = 0; k < 8; ++k) {
      aH[k]  = ldA_bf(hrow,  k * 32, lane);
      aH0[k] = ldA_bf(h0row, k * 32, lane);
    }
#pragma unroll
    for (int mt = 0; mt < 4; ++mt) {
      const int mc0 = w * 64 + mt * 16;
      const bf16_t* wir = wi + (size_t)(mc0 + lrow) * 512;
      const bf16_t* wjr = wj + (size_t)(mc0 + lrow) * DD;
      v8f aG = v8f_splat(0.0f), aV = v8f_splat(0.0f);
#pragma unroll
      for (int k = 0; k < 8; ++k) aG = wmma_bf(aH[k],  ldB_bf(wir,       k * 32, lane), aG);
#pragma unroll
      for (int k = 0; k < 8; ++k) aG = wmma_bf(aH0[k], ldB_bf(wir, 256 + k * 32, lane), aG);
#pragma unroll
      for (int k = 0; k < 8; ++k) aV = wmma_bf(aH[k],  ldB_bf(wjr,       k * 32, lane), aV);
      const int col = mc0 + lrow;
      const float gb = ib[col], vb = jb[col];
      float acc = 0.0f;
#pragma unroll
      for (int v = 0; v < 8; ++v) acc += sigmoidf_(aG[v] + gb) * (aV[v] + vb);
      psum[mt] += acc;
    }
  }

  // lanes L and L+16 hold the same column (rows 0-7 vs 8-15): xor-16 reduce.
#pragma unroll
  for (int mt = 0; mt < 4; ++mt) {
    float tot = psum[mt] + __shfl_xor(psum[mt], 16, 32);
    if (lane < 16) out[(size_t)b * MM + w * 64 + mt * 16 + lane] = tot;
  }
}

// ---------------------------------------------------------------------------
extern "C" void kernel_launch(void* const* d_in, const int* in_sizes, int n_in,
                              void* d_out, int out_size, void* d_ws, size_t ws_size,
                              hipStream_t stream) {
  const int*   atom  = (const int*)d_in[0];
  const float* adj   = (const float*)d_in[1];
  const float* embed = (const float*)d_in[2];
  const float* msgW  = (const float*)d_in[3];
  const float* msgb  = (const float*)d_in[4];
  const float* Wih   = (const float*)d_in[5];
  const float* bih   = (const float*)d_in[6];
  const float* Whh   = (const float*)d_in[7];
  const float* bhh   = (const float*)d_in[8];
  const float* iW    = (const float*)d_in[9];
  const float* ib    = (const float*)d_in[10];
  const float* jW    = (const float*)d_in[11];
  const float* jb    = (const float*)d_in[12];
  float* out = (float*)d_out;

  // workspace carve-out (~272 MB total, 256B aligned slices)
  char* ws = (char*)d_ws;
  size_t off = 0;
  auto take = [&](size_t bytes) -> char* {
    char* p = ws + off;
    off = (off + bytes + 255) & ~(size_t)255;
    return p;
  };
  bf16_t* wbMsg = (bf16_t*)take((size_t)3 * 1024 * 256 * 2);
  bf16_t* wbIh  = (bf16_t*)take((size_t)768 * 256 * 2);
  bf16_t* wbHh  = (bf16_t*)take((size_t)768 * 256 * 2);
  bf16_t* wbI   = (bf16_t*)take((size_t)512 * 512 * 2);
  bf16_t* wbJ   = (bf16_t*)take((size_t)512 * 256 * 2);
  bf16_t* h0b   = (bf16_t*)take((size_t)ROWS * DD * 2);
  bf16_t* hA    = (bf16_t*)take((size_t)ROWS * DD * 2);
  bf16_t* hB    = (bf16_t*)take((size_t)ROWS * DD * 2);
  bf16_t* msum  = (bf16_t*)take((size_t)ROWS * DD * 2);

  k_convert<<<(3 * 1024 * 256 + 255) / 256, 256, 0, stream>>>(msgW, wbMsg, 3 * 1024 * 256);
  k_convert<<<(768 * 256 + 255) / 256, 256, 0, stream>>>(Wih, wbIh, 768 * 256);
  k_convert<<<(768 * 256 + 255) / 256, 256, 0, stream>>>(Whh, wbHh, 768 * 256);
  k_convert<<<(512 * 512 + 255) / 256, 256, 0, stream>>>(iW, wbI, 512 * 512);
  k_convert<<<(512 * 256 + 255) / 256, 256, 0, stream>>>(jW, wbJ, 512 * 256);

  k_embed<<<(ROWS * DD) / 256, 256, 0, stream>>>(atom, embed, hA, h0b);

  bf16_t* hc = hA;
  bf16_t* hn = hB;
  for (int step = 0; step < 3; ++step) {
    k_msg<<<MBATCH, 256, 0, stream>>>(hc, adj, wbMsg + (size_t)step * 1024 * 256,
                                      msgb + step * 1024, msum);
    k_gru<<<ROWS / 16, 256, 0, stream>>>(msum, hc, wbIh, bih, wbHh, bhh, hn,
                                         step > 0 ? 1 : 0);
    bf16_t* t = hc; hc = hn; hn = t;
  }
  k_readout<<<MBATCH, 256, 0, stream>>>(hc, h0b, wbI, ib, wbJ, jb, out);
}
